// EpisodicMemory_75342316306643
// MI455X (gfx1250) — matrix operand order repaired and checked by reading, not compiled, not run on hardware
//
#include <hip/hip_runtime.h>
#include <math.h>

// ---------------------------------------------------------------------------
// EpisodicMemory reward on MI455X (gfx1250, wave32).
//
// One streaming pass over memory[M,32] (256 MB -> HBM-bound, ~11us floor).
//   sq[m,q] = max(q2[q] + m2[m] - 2*dot(mem[m], query[q]), 0)
// -2*dot via v_wmma_f32_16x16x32_bf16 with f32 -> (bf16 hi + bf16 lo) split
// (the -2 is folded into the query operands; exact power-of-2 scale).
// q2 + m2 is injected by ONE v_wmma_f32_16x16x4_f32 rank-2 update whose K
// layout (lanes 0-15 -> K0/K1, lanes 16-31 -> K2/K3) sums the per-lane-half
// partial sums of squares -- so the hot loop needs NO cross-lane shuffles.
// Selection is done on sq (kdist = eps/(sq/mean+eps) is monotone decreasing
// in sq, so the k smallest kdist == kdist of the k largest sq; the mean is
// only needed in the tiny finalize step).
// ---------------------------------------------------------------------------

typedef __attribute__((ext_vector_type(16))) __bf16 v16bf;
typedef __attribute__((ext_vector_type(8)))  float  v8f;
typedef __attribute__((ext_vector_type(2)))  float  v2f;
typedef __attribute__((ext_vector_type(4)))  float  f32x4;

#define KK        10      // top-k (reference k)
#define QTILES    4       // 64 queries / 16
#define WAVES_PB  8       // 256 threads / 32

union BF16x16 { v16bf v; unsigned w[8]; };

// Split two f32 into packed bf16 hi words and bf16 lo (residual) words.
// hi = truncation (residual exact by Sterbenz); lo = truncation of residual
// (error ~2^-16 relative -- ample for top-k selection on sq ~ 64).
// v_perm_b32 packs 2 bf16 per instruction.
__device__ __forceinline__ void bf16_split2(float fa, float fb,
                                            unsigned& hw, unsigned& lw) {
    unsigned ua = __float_as_uint(fa), ub = __float_as_uint(fb);
    float fla = fa - __uint_as_float(ua & 0xFFFF0000u);
    float flb = fb - __uint_as_float(ub & 0xFFFF0000u);
    hw = __builtin_amdgcn_perm(ub, ua, 0x07060302u);                    // {hi(fb),hi(fa)}
    lw = __builtin_amdgcn_perm(__float_as_uint(flb), __float_as_uint(fla),
                               0x07060302u);                             // {hi(flb),hi(fla)}
}

// Branchless insert of x into ascending sorted top-K list (lst[0] = min).
__device__ __forceinline__ void topk_insert(float (&lst)[KK], float x) {
    float c = x;
#pragma unroll
    for (int i = KK - 1; i >= 0; --i) {
        float mx = fmaxf(c, lst[i]);
        c        = fminf(c, lst[i]);
        lst[i]   = mx;
    }
}

// ---------------------------------------------------------------------------
// Kernel 0: zero the global f64 sum accumulator.
// ---------------------------------------------------------------------------
__global__ void episodic_zero(double* s) { *s = 0.0; }

// ---------------------------------------------------------------------------
// Kernel 1: streaming distance pass. Each wave grid-strides over 16-row
// memory tiles; 16 WMMAs/tile produce sq for a 16x64 block; epilogue keeps
// per-lane top-10 lists and the global sum of sq.
// ---------------------------------------------------------------------------
__global__ __launch_bounds__(256) void episodic_main(
    const float* __restrict__ query,   // [64, 32]
    const float* __restrict__ mem,     // [M, 32]
    float*       __restrict__ topbuf,  // [nwaves, 64, KK]
    double*      __restrict__ gsum,
    int ntiles, int nwaves)
{
    const int lane = threadIdx.x & 31;
    const int wid  = blockIdx.x * WAVES_PB + (threadIdx.x >> 5);
    const int hi   = lane >> 4;        // 0: low K half, 1: high K half
    const int l16  = lane & 15;

    // ---- Queries -> B operands (hi/lo bf16, scaled by -2) + q2 partials ---
    // bf16 B layout used here (32x16, contiguous K): lanes 0-15 hold K=0..15 of col N=lane,
    //                        lanes 16-31 hold K=16..31 of col N=lane-16.
    BF16x16 bhi[QTILES], blo[QTILES];
    v2f     bfix[QTILES];              // f32 4x16 B: rows {1, q2_partial, 1, q2_partial}
#pragma unroll
    for (int t = 0; t < QTILES; ++t) {
        const f32x4* qp4 = (const f32x4*)(query + (size_t)(t * 16 + l16) * 32);
        f32x4 g[4];
#pragma unroll
        for (int c = 0; c < 4; ++c) g[c] = qp4[hi * 4 + c];   // dims 16*hi .. +15
        float qs = 0.0f;
#pragma unroll
        for (int p = 0; p < 8; ++p) {
            float fa = g[(2 * p) >> 2][(2 * p) & 3];
            float fb = g[(2 * p + 1) >> 2][(2 * p + 1) & 3];
            qs = fmaf(fa, fa, qs);
            qs = fmaf(fb, fb, qs);
            bf16_split2(-2.0f * fa, -2.0f * fb, bhi[t].w[p], blo[t].w[p]);
        }
        bfix[t][0] = 1.0f;   // K0 (lanes<16) / K2 (lanes>=16) row of ones
        bfix[t][1] = qs;     // K1 / K3 row: per-half partial q2 of this column
    }

    // ---- Per-lane top-10 lists (sq values, ascending; lst[0] = min) -------
    float lst[QTILES][KK];
#pragma unroll
    for (int t = 0; t < QTILES; ++t)
#pragma unroll
        for (int i = 0; i < KK; ++i) lst[t][i] = -1.0f;

    float lsum = 0.0f;
    const v8f zc = {0.f, 0.f, 0.f, 0.f, 0.f, 0.f, 0.f, 0.f};

    for (int tile = wid; tile < ntiles; tile += nwaves) {
        // bf16 A layout used here (16x32, contiguous K, same as B): row = lane%16;
        //   lanes 0-15 : elems 0-15 = K 0-15
        //   lanes 16-31: elems 0-15 = K 16-31
        const f32x4* rp4 = (const f32x4*)(mem + ((size_t)tile * 16 + l16) * 32);
        f32x4 f[4];
        f[0] = __builtin_nontemporal_load(rp4 + hi * 4 + 0);
        f[1] = __builtin_nontemporal_load(rp4 + hi * 4 + 1);
        f[2] = __builtin_nontemporal_load(rp4 + hi * 4 + 2);
        f[3] = __builtin_nontemporal_load(rp4 + hi * 4 + 3);
        if (tile + nwaves < ntiles)   // global_prefetch_b8 for our next tile
            __builtin_prefetch(mem + ((size_t)(tile + nwaves) * 16 + l16) * 32, 0, 1);

        BF16x16 ahi, alo;
        float ms = 0.0f;
#pragma unroll
        for (int p = 0; p < 8; ++p) {
            float fa = f[(2 * p) >> 2][(2 * p) & 3];
            float fb = f[(2 * p + 1) >> 2][(2 * p + 1) & 3];
            ms = fmaf(fa, fa, ms);
            ms = fmaf(fb, fb, ms);
            bf16_split2(fa, fb, ahi.w[p], alo.w[p]);
        }
        // f32 16x4 A operand: K0/K2 = per-half partial m2 of this row, K1/K3 = 1.
        v2f afix;
        afix[0] = ms;
        afix[1] = 1.0f;

        // acc = (m2[m] + q2[q]) - 2*dot(mem[m], query[q])  -- 4 WMMAs per tile.
        v8f acc[QTILES];
#pragma unroll
        for (int t = 0; t < QTILES; ++t) {
            acc[t] = __builtin_amdgcn_wmma_f32_16x16x4_f32(
                false, afix, false, bfix[t], (short)0, zc, false, false);
            acc[t] = __builtin_amdgcn_wmma_f32_16x16x32_bf16(
                false, ahi.v, false, bhi[t].v, (short)0, acc[t], false, false);
            acc[t] = __builtin_amdgcn_wmma_f32_16x16x32_bf16(
                false, ahi.v, false, blo[t].v, (short)0, acc[t], false, false);
            acc[t] = __builtin_amdgcn_wmma_f32_16x16x32_bf16(
                false, alo.v, false, bhi[t].v, (short)0, acc[t], false, false);
        }

        // ---- Epilogue: clamp, global sum, per-lane top-k -------------------
        // C layout: elem j -> row (j + 8*hi), col (lane%16) of query tile t.
#pragma unroll
        for (int j = 0; j < 8; ++j)
#pragma unroll
            for (int t = 0; t < QTILES; ++t) {
                float d = fmaxf(acc[t][j], 0.0f);
                lsum += d;
                if (d > lst[t][0]) topk_insert(lst[t], d);  // rare after warmup
            }
    }

    // ---- Merge partner lane (lane+16) lists into lanes 0-15 ---------------
#pragma unroll
    for (int t = 0; t < QTILES; ++t)
#pragma unroll
        for (int i = 0; i < KK; ++i) {
            float o = __shfl(lst[t][i], lane + 16, 32);  // wraps for hi lanes
            if ((lane < 16) && (o > lst[t][0])) topk_insert(lst[t], o);
        }

    // ---- Emit per-wave results --------------------------------------------
    if (lane < 16) {
#pragma unroll
        for (int t = 0; t < QTILES; ++t) {
            float* dst = topbuf + ((size_t)wid * 64 + (t * 16 + lane)) * KK;
#pragma unroll
            for (int i = 0; i < KK; ++i) dst[i] = lst[t][i];
        }
    }

    // Wave-reduce the sq sum, one f64 atomic per wave.
#pragma unroll
    for (int off = 16; off >= 1; off >>= 1) lsum += __shfl_xor(lsum, off, 32);
    if (lane == 0) atomicAdd(gsum, (double)lsum);
}

// ---------------------------------------------------------------------------
// Kernel 2: per query, merge all per-wave top-10 lists (log2 LDS tree),
// apply the kernel function with the global mean, emit 1/sqrt(sum + c).
// ---------------------------------------------------------------------------
__global__ __launch_bounds__(256) void episodic_finalize(
    const float*  __restrict__ topbuf,
    const double* __restrict__ gsum,
    float*        __restrict__ out,
    int nwaves, long long count)
{
    __shared__ float sh[256 * KK];
    const int q   = blockIdx.x;      // 0..63
    const int tid = threadIdx.x;

    float lst[KK];
#pragma unroll
    for (int i = 0; i < KK; ++i) lst[i] = -1.0f;

    for (int w = tid; w < nwaves; w += blockDim.x) {
        const float* p = topbuf + ((size_t)w * 64 + q) * KK;
#pragma unroll
        for (int i = 0; i < KK; ++i) {
            float v = p[i];
            if (v > lst[0]) topk_insert(lst, v);
        }
    }

    // Tree-merge 256 sorted lists -> 1.
    for (int half = 128; half >= 1; half >>= 1) {
        if (tid >= half && tid < 2 * half) {
#pragma unroll
            for (int i = 0; i < KK; ++i) sh[tid * KK + i] = lst[i];
        }
        __syncthreads();
        if (tid < half) {
            const float* b = &sh[(tid + half) * KK];
#pragma unroll
            for (int i = 0; i < KK; ++i) {
                float v = b[i];
                if (v > lst[0]) topk_insert(lst, v);
            }
        }
        __syncthreads();
    }

    if (tid == 0) {
        float mean = (float)(*gsum / (double)count);
        float ks = 0.0f;
#pragma unroll
        for (int i = 0; i < KK; ++i)
            ks += 1e-5f / (lst[i] / mean + 1e-5f);   // kdist of the k largest sq
        out[q] = 1.0f / sqrtf(ks + 1e-5f);
    }
}

// ---------------------------------------------------------------------------
extern "C" void kernel_launch(void* const* d_in, const int* in_sizes, int n_in,
                              void* d_out, int out_size, void* d_ws, size_t ws_size,
                              hipStream_t stream) {
    const float* query  = (const float*)d_in[0];   // [64, 32] fp32
    const float* memory = (const float*)d_in[1];   // [M, 32]  fp32
    float*       out    = (float*)d_out;           // [64]     fp32

    const int M      = in_sizes[1] / 32;
    const int ntiles = M / 16;                     // M = 2,000,000 -> 125,000

    // Workspace: [0,8) f64 sum; top lists at offset 256.
    double* d_sum = (double*)d_ws;
    float*  d_top = (float*)((char*)d_ws + 256);

    // Deterministic grid sizing from ws_size: 64 queries * KK floats per wave.
    const size_t per_wave = (size_t)64 * KK * sizeof(float);
    size_t max_waves = (ws_size > 256) ? (ws_size - 256) / per_wave : (size_t)WAVES_PB;
    int nwg = (int)(max_waves / WAVES_PB);
    if (nwg > 256) nwg = 256;                      // 2048 waves, ~5.2 MB scratch
    if (nwg < 1)  nwg = 1;
    const int nwaves = nwg * WAVES_PB;

    episodic_zero<<<1, 1, 0, stream>>>(d_sum);
    episodic_main<<<nwg, 256, 0, stream>>>(query, memory, d_top, d_sum, ntiles, nwaves);
    episodic_finalize<<<out_size, 256, 0, stream>>>(d_top, d_sum, out, nwaves,
                                                    (long long)(in_sizes[0] / 32) * M);
}
